// ResGraphConvUnpool_30279519436851
// MI455X (gfx1250) — matrix operand order repaired and checked by reading, not compiled, not run on hardware
//
#include <hip/hip_runtime.h>

#define Bq 4
#define Nq 4096
#define Dq 128
#define BN (Bq * Nq)
#define Kq 8
#define NB 12

typedef __attribute__((ext_vector_type(16))) _Float16 v16h;
typedef __attribute__((ext_vector_type(8)))  float    v8f;
typedef __attribute__((ext_vector_type(4)))  unsigned int u4;

struct __align__(8) h4 { _Float16 x, y, z, w; };

// ---- WMMA fragment loaders (per CDNA5 ISA 7.12.2 VGPR layouts) ----
// A 16x32 f16: lane<16 -> K {0..7,16..23}; lane>=16 -> K {8..15,24..31}
__device__ __forceinline__ v16h load_a(const _Float16* p) {
  union { u4 u[2]; v16h h; } t;
  t.u[0] = *(const u4*)p;          // K base .. +7
  t.u[1] = *(const u4*)(p + 16);   // K base+16 .. +23
  return t.h;
}
// B 32x16 f16: lane<16 -> K {0..7,16..23} of column (lane); lane>=16 -> K {8..15,24..31}
__device__ __forceinline__ v16h load_b(const _Float16* p) {
  union { u4 u[2]; v16h h; } t;
  t.u[0] = *(const u4*)p;
  t.u[1] = *(const u4*)(p + 16);
  return t.h;
}

// ---- kNN: one thread per query, xyz for the batch staged in LDS ----
__global__ void knn_kernel(const float* __restrict__ xyz, int* __restrict__ idx) {
  __shared__ float sx[Nq], sy[Nq], sz[Nq];   // 48 KB
  const int b  = blockIdx.x >> 4;            // 16 blocks per batch
  const int qb = blockIdx.x & 15;
  const float* base = xyz + (size_t)b * 3 * Nq;
  for (int t = threadIdx.x; t < Nq; t += 256) {
    sx[t] = base[t];
    sy[t] = base[Nq + t];
    sz[t] = base[2 * Nq + t];
  }
  __syncthreads();
  const int q = qb * 256 + threadIdx.x;
  const float qx = sx[q], qy = sy[q], qz = sz[q];
  float bd[8]; int bi[8];
#pragma unroll
  for (int j = 0; j < 8; ++j) { bd[j] = 3.0e38f; bi[j] = 0; }
  for (int m = 0; m < Nq; ++m) {
    const float dx = qx - sx[m], dy = qy - sy[m], dz = qz - sz[m];
    const float d = dx * dx + dy * dy + dz * dz;
    if (d < bd[7]) {                 // strict '<' keeps earlier index on ties (matches top_k)
      bd[7] = d; bi[7] = m;
#pragma unroll
      for (int j = 7; j > 0; --j) {  // fully unrolled -> arrays stay in VGPRs
        if (bd[j] < bd[j - 1]) {
          float td = bd[j]; bd[j] = bd[j - 1]; bd[j - 1] = td;
          int   ti = bi[j]; bi[j] = bi[j - 1]; bi[j - 1] = ti;
        }
      }
    }
  }
  int* outp = idx + ((size_t)b * Nq + q) * Kq;
#pragma unroll
  for (int j = 0; j < 8; ++j) outp[j] = bi[j];
}

// ---- (B,D,N) -> (B,N,D) tiled transpose ----
__global__ void transpose_in(const float* __restrict__ in, float* __restrict__ X) {
  __shared__ float tile[32][33];
  const int n0 = blockIdx.x * 32, d0 = blockIdx.y * 32, b = blockIdx.z;
  for (int yy = threadIdx.y; yy < 32; yy += 8)
    tile[yy][threadIdx.x] = in[((size_t)b * Dq + d0 + yy) * Nq + n0 + threadIdx.x];
  __syncthreads();
  for (int yy = threadIdx.y; yy < 32; yy += 8)
    X[((size_t)b * Nq + n0 + yy) * Dq + d0 + threadIdx.x] = tile[threadIdx.x][yy];
}

// ---- (B,N,D) -> (B,D,N) for the final points output ----
__global__ void transpose_out(const float* __restrict__ X, float* __restrict__ outp) {
  __shared__ float tile[32][33];
  const int n0 = blockIdx.x * 32, d0 = blockIdx.y * 32, b = blockIdx.z;
  for (int yy = threadIdx.y; yy < 32; yy += 8)
    tile[yy][threadIdx.x] = X[((size_t)b * Nq + n0 + yy) * Dq + d0 + threadIdx.x];
  __syncthreads();
  for (int yy = threadIdx.y; yy < 32; yy += 8)
    outp[((size_t)b * Dq + d0 + yy) * Nq + n0 + threadIdx.x] = tile[threadIdx.x][yy];
}

// ---- f32 -> f16 weight conversion (done once per launch) ----
__global__ void convert_w(const float* __restrict__ w1, const float* __restrict__ w2,
                          _Float16* __restrict__ o1, _Float16* __restrict__ o2, int n) {
  const int i = blockIdx.x * 256 + threadIdx.x;
  if (i < n) { o1[i] = (_Float16)w1[i]; o2[i] = (_Float16)w2[i]; }
}

// ---- per-block prep: P = relu(X) (f16), S = sum_k relu(X[neighbor_k]) (f16) ----
// one wave per point-row (128 channels = 32 lanes x float4)
__global__ void prep_kernel(const float* __restrict__ X, const int* __restrict__ idx,
                            _Float16* __restrict__ P, _Float16* __restrict__ S) {
  const int wave = threadIdx.x >> 5;
  const int lane = threadIdx.x & 31;
  const int row  = blockIdx.x * 8 + wave;        // [0, BN)
  const int b    = row >> 12;                    // row / Nq
  const float4 v = ((const float4*)(X + (size_t)row * Dq))[lane];
  const float px = fmaxf(v.x, 0.f), py = fmaxf(v.y, 0.f);
  const float pz = fmaxf(v.z, 0.f), pw = fmaxf(v.w, 0.f);
  float sx = 0.f, sy = 0.f, sz = 0.f, sw = 0.f;
  const int* nb = idx + (size_t)row * Kq;
#pragma unroll
  for (int k = 0; k < Kq; ++k) {
    const int nr = b * Nq + nb[k];
    const float4 g = ((const float4*)(X + (size_t)nr * Dq))[lane];
    sx += fmaxf(g.x, 0.f); sy += fmaxf(g.y, 0.f);
    sz += fmaxf(g.z, 0.f); sw += fmaxf(g.w, 0.f);
  }
  h4 hp; hp.x = (_Float16)px; hp.y = (_Float16)py; hp.z = (_Float16)pz; hp.w = (_Float16)pw;
  h4 hs; hs.x = (_Float16)sx; hs.y = (_Float16)sy; hs.z = (_Float16)sz; hs.w = (_Float16)sw;
  ((h4*)(P + (size_t)row * Dq))[lane] = hp;
  ((h4*)(S + (size_t)row * Dq))[lane] = hs;
}

// ---- WMMA GEMM: X[n,o] = (sum_i W1[o,i]P[n,i] + W2[o,i]S[n,i]) / 9 + X[n,o] ----
// 8 waves/block, each wave owns a 16-row strip, loops over 8 output-channel tiles.
__global__ void __launch_bounds__(256) gemm_kernel(const _Float16* __restrict__ P,
                                                   const _Float16* __restrict__ S,
                                                   const _Float16* __restrict__ W1,
                                                   const _Float16* __restrict__ W2,
                                                   float* __restrict__ X) {
  const int wave = threadIdx.x >> 5;
  const int lane = threadIdx.x & 31;
  const int row0 = blockIdx.x * 128 + wave * 16;
  const int mrow = lane & 15;
  const int kA = (lane & 16) ? 8 : 0;    // A-matrix K split per ISA layout
  const int kB = (lane & 16) ? 8 : 0;    // B-matrix K split per ISA layout

  v16h aP[4], aS[4];
  const _Float16* prow = P + (size_t)(row0 + mrow) * Dq + kA;
  const _Float16* srow = S + (size_t)(row0 + mrow) * Dq + kA;
#pragma unroll
  for (int ks = 0; ks < 4; ++ks) {       // K = 128 in 4 steps of 32
    aP[ks] = load_a(prow + 32 * ks);
    aS[ks] = load_a(srow + 32 * ks);
  }
  const float inv = 1.0f / 9.0f;
#pragma unroll
  for (int ot = 0; ot < 8; ++ot) {
    const int o0 = ot * 16;
    const _Float16* w1row = W1 + (size_t)(o0 + mrow) * Dq + kB;
    const _Float16* w2row = W2 + (size_t)(o0 + mrow) * Dq + kB;
    v8f acc = {0.f, 0.f, 0.f, 0.f, 0.f, 0.f, 0.f, 0.f};
#pragma unroll
    for (int ks = 0; ks < 4; ++ks) {
      const v16h b1 = load_b(w1row + 32 * ks);
      const v16h b2 = load_b(w2row + 32 * ks);
      acc = __builtin_amdgcn_wmma_f32_16x16x32_f16(false, aP[ks], false, b1,
                                                   (short)0, acc, false, false);
      acc = __builtin_amdgcn_wmma_f32_16x16x32_f16(false, aS[ks], false, b2,
                                                   (short)0, acc, false, false);
    }
    // C/D layout: lane -> column (lane&15), VGPR r -> row r + 8*(lane>=16)
    const int rbase = row0 + ((lane >> 4) << 3);
    const int col   = o0 + (lane & 15);
#pragma unroll
    for (int r = 0; r < 8; ++r) {
      float* xp = X + (size_t)(rbase + r) * Dq + col;
      *xp = acc[r] * inv + *xp;          // fused mean + residual, in place (own elements only)
    }
  }
}

// ---- last-block 6-channel head: new_xyz = (Wc@p + Wn@S)/9 reshaped + xyz ----
__global__ void xyz_kernel(const _Float16* __restrict__ P, const _Float16* __restrict__ S,
                           const float* __restrict__ wc, const float* __restrict__ wn,
                           const float* __restrict__ xyz, float* __restrict__ out0) {
  const int t = blockIdx.x * 256 + threadIdx.x;   // [0, BN)
  const int b = t >> 12;
  const int n = t & (Nq - 1);
  const _Float16* pr = P + (size_t)t * Dq;
  const _Float16* sr = S + (size_t)t * Dq;
  float acc[6] = {0.f, 0.f, 0.f, 0.f, 0.f, 0.f};
  for (int i = 0; i < Dq; ++i) {
    const float pv = (float)pr[i];
    const float sv = (float)sr[i];
#pragma unroll
    for (int o = 0; o < 6; ++o)
      acc[o] += wc[o * Dq + i] * pv + wn[o * Dq + i] * sv;
  }
  const float inv = 1.0f / 9.0f;
#pragma unroll
  for (int c = 0; c < 3; ++c) {
    const float base = xyz[((size_t)b * 3 + c) * Nq + n];
#pragma unroll
    for (int j = 0; j < 2; ++j)
      out0[(((size_t)b * 3 + c) * 2 + j) * Nq + n] = acc[2 * c + j] * inv + base;
  }
}

extern "C" void kernel_launch(void* const* d_in, const int* in_sizes, int n_in,
                              void* d_out, int out_size, void* d_ws, size_t ws_size,
                              hipStream_t stream) {
  (void)in_sizes; (void)n_in; (void)out_size; (void)ws_size;
  const float* xyz    = (const float*)d_in[0];
  const float* points = (const float*)d_in[1];
  const float* cw1    = (const float*)d_in[2];
  const float* cw2    = (const float*)d_in[3];
  const float* wc     = (const float*)d_in[4];
  const float* wn     = (const float*)d_in[5];
  float* out = (float*)d_out;

  // workspace layout (bytes), ~17.3 MB total
  char* ws = (char*)d_ws;
  float*    X    = (float*)(ws);                  // (B,N,D) f32 residual stream, 8 MB
  _Float16* P    = (_Float16*)(ws + 8388608);     // relu(X) f16, 4 MB
  _Float16* S    = (_Float16*)(ws + 12582912);    // neighbor-sum f16, 4 MB
  int*      idxb = (int*)(ws + 16777216);         // kNN indices, 512 KB
  _Float16* W1h  = (_Float16*)(ws + 17301504);    // conv_w1 f16, 384 KB
  _Float16* W2h  = (_Float16*)(ws + 17694720);    // conv_w2 f16, 384 KB

  knn_kernel<<<Bq * (Nq / 256), 256, 0, stream>>>(xyz, idxb);
  transpose_in<<<dim3(Nq / 32, Dq / 32, Bq), dim3(32, 8), 0, stream>>>(points, X);
  const int nw = NB * Dq * Dq;
  convert_w<<<(nw + 255) / 256, 256, 0, stream>>>(cw1, cw2, W1h, W2h, nw);

  for (int i = 0; i < NB; ++i) {
    prep_kernel<<<BN / 8, 256, 0, stream>>>(X, idxb, P, S);
    if (i == NB - 1)  // head uses this block's P,S (pre-update activations)
      xyz_kernel<<<BN / 256, 256, 0, stream>>>(P, S, wc, wn, xyz, out);
    gemm_kernel<<<BN / 128, 256, 0, stream>>>(P, S, W1h + (size_t)i * Dq * Dq,
                                              W2h + (size_t)i * Dq * Dq, X);
  }
  transpose_out<<<dim3(Nq / 32, Dq / 32, Bq), dim3(32, 8), 0, stream>>>(
      X, out + (size_t)Bq * 3 * 2 * Nq);
}
